// DeformConv1d_45827301048698
// MI455X (gfx1250) — matrix operand order repaired and checked by reading, not compiled, not run on hardware
//
#include <hip/hip_runtime.h>

#define K_TAPS  3
#define CIN     256
#define COUT    256
#define LIN     4096
#define OUT_LEN 4094
#define BT      64           // t-tile rows per block
#define AROW    40           // padded LDS row stride (bf16 elems, 80 bytes)
#define BROW    40
#define NCHUNK  24           // 3 taps * 8 chunks of 32 channels
#define WTILE_BYTES (NCHUNK * COUT * 32 * 2)   // 393216 B of pre-tiled bf16 weights

typedef __attribute__((ext_vector_type(16))) __bf16 v16bf;
typedef __attribute__((ext_vector_type(8)))  float  v8f;

union Frag { uint4 q[2]; v16bf v; };

__device__ __forceinline__ unsigned short f32_to_bf16(float f) {
  union { float f; unsigned u; } c; c.f = f;
  unsigned u = c.u + 0x7FFFu + ((c.u >> 16) & 1u);  // round-to-nearest-even
  return (unsigned short)(u >> 16);
}

// Pre-convert weights to bf16, tiled per reduction chunk m = k*8 + cc:
//   wb[((m*256 + o)*32) + cl] = bf16(weight[o, cc*32 + cl, k])
__global__ __launch_bounds__(256) void convert_weight_tiles(
    const float* __restrict__ w, unsigned short* __restrict__ wb)
{
  const int idx = blockIdx.x * 256 + threadIdx.x;   // 24576 threads, 8 elems each
  const int cl8 = idx & 3;
  const int o   = (idx >> 2) & 255;
  const int cc  = (idx >> 10) & 7;
  const int k   = idx >> 13;
  if (k >= K_TAPS) return;
  unsigned short pk[8];
  #pragma unroll
  for (int j = 0; j < 8; ++j) {
    const int c = cc * 32 + cl8 * 8 + j;
    pk[j] = f32_to_bf16(w[((size_t)o * CIN + c) * K_TAPS + k]);
  }
  uint4 q;
  q.x = pk[0] | ((unsigned)pk[1] << 16);
  q.y = pk[2] | ((unsigned)pk[3] << 16);
  q.z = pk[4] | ((unsigned)pk[5] << 16);
  q.w = pk[6] | ((unsigned)pk[7] << 16);
  *(uint4*)&wb[(size_t)idx * 8] = q;
}

__global__ __launch_bounds__(256) void deform_conv1d_wmma(
    const float* __restrict__ x, const float* __restrict__ offsets,
    const float* __restrict__ weight, const float* __restrict__ bias,
    const unsigned short* __restrict__ wtiles, const int use_ws,
    float* __restrict__ out)
{
  __shared__ __align__(16) unsigned short As[2][BT * AROW];     // 2 x  5120 B
  __shared__ __align__(16) unsigned short Bs[2][COUT * BROW];   // 2 x 20480 B
  __shared__ float Ct[COUT * (BT + 1)];                         //     66560 B

  const int tid = threadIdx.x;
  const int b   = blockIdx.z;
  const int t0  = blockIdx.x * BT;

  // wave decomposition: ts = t-subtile (16 rows), og = o-half (8 subtiles each)
  const int wid  = tid >> 5;
  const int lane = tid & 31;
  const int ts   = wid & 3;
  const int og   = wid >> 2;
  const int lhi  = (lane >= 16);
  const int l16  = lane & 15;

  v8f acc[8] = {};

  // A-staging roles: 64 t-rows x 4 channel-groups of 8
  const int a_t   = tid & 63;
  const int a_cs  = tid >> 6;
  const int tglob = t0 + a_t;
  const int tcl   = tglob < OUT_LEN ? tglob : OUT_LEN - 1;

  const float* xb = x + (size_t)b * CIN * LIN;

  // hoist per-(t,k) interpolation params: T = clip(t + k + off, t, t+2)
  int   u0k[K_TAPS];
  float w0k[K_TAPS], w1k[K_TAPS];
  #pragma unroll
  for (int k = 0; k < K_TAPS; ++k) {
    float off  = offsets[((size_t)b * OUT_LEN + tcl) * K_TAPS + k];
    float base = (float)tcl;
    float T    = fminf(fmaxf(base + (float)k + off, base), base + 2.0f);
    int u0 = (int)T;                       // T >= 0 so trunc == floor
    if (u0 > LIN - 2) u0 = LIN - 2;
    float d = T - (float)u0;
    u0k[k] = u0; w0k[k] = 1.0f - d; w1k[k] = d;
  }

  auto stageA = [&](int buf, int m) {
    const int kk = m >> 3, cc = m & 7;
    int u0 = u0k[0]; float w0 = w0k[0], w1 = w1k[0];
    if (kk == 1) { u0 = u0k[1]; w0 = w0k[1]; w1 = w1k[1]; }
    else if (kk == 2) { u0 = u0k[2]; w0 = w0k[2]; w1 = w1k[2]; }
    const float* p = xb + u0 + (size_t)(cc * 32 + a_cs * 8) * LIN;
    unsigned short pk[8];
    #pragma unroll
    for (int j = 0; j < 8; ++j) {
      float v = w0 * p[0] + w1 * p[1];
      pk[j] = f32_to_bf16(v);
      p += LIN;
    }
    uint4 q;
    q.x = pk[0] | ((unsigned)pk[1] << 16);
    q.y = pk[2] | ((unsigned)pk[3] << 16);
    q.z = pk[4] | ((unsigned)pk[5] << 16);
    q.w = pk[6] | ((unsigned)pk[7] << 16);
    *(uint4*)&As[buf][a_t * AROW + a_cs * 8] = q;
  };

  auto stageB = [&](int buf, int m) {
    if (use_ws) {
      // pre-tiled bf16 chunk: contiguous 16 KB -> async b128 copies into LDS rows
      const unsigned short* src = wtiles + (size_t)m * (COUT * 32);
      const int q = tid & 3, ob = tid >> 2;
      #pragma unroll
      for (int it = 0; it < 4; ++it) {
        const int o = ob + 64 * it;
        unsigned long long ga = (unsigned long long)(const void*)(src + o * 32 + q * 8);
        unsigned la = (unsigned)(size_t)(const void*)&Bs[buf][0]
                    + (unsigned)(o * (BROW * 2) + q * 16);
        asm volatile("global_load_async_to_lds_b128 %0, %1, off"
                     :: "v"(la), "v"(ga) : "memory");
      }
    } else {
      // fallback: gather + convert from original f32 weight [o, c, k]
      const int kk = m >> 3, cc = m & 7;
      const float* wp = weight + (size_t)tid * (CIN * K_TAPS)
                               + (size_t)(cc * 32) * K_TAPS + kk;
      #pragma unroll
      for (int h = 0; h < 4; ++h) {
        unsigned short pk[8];
        #pragma unroll
        for (int j = 0; j < 8; ++j) pk[j] = f32_to_bf16(wp[(h * 8 + j) * K_TAPS]);
        uint4 q;
        q.x = pk[0] | ((unsigned)pk[1] << 16);
        q.y = pk[2] | ((unsigned)pk[3] << 16);
        q.z = pk[4] | ((unsigned)pk[5] << 16);
        q.w = pk[6] | ((unsigned)pk[7] << 16);
        *(uint4*)&Bs[buf][tid * BROW + h * 8] = q;
      }
    }
  };

  // ---- software pipeline: stage chunk 0, then overlap stage(m+1) with wmma(m)
  stageA(0, 0);
  stageB(0, 0);

  for (int m = 0; m < NCHUNK; ++m) {
    const int buf = m & 1;
    if (use_ws) asm volatile("s_wait_asynccnt 0x0" ::: "memory");
    __syncthreads();

    if (m + 1 < NCHUNK) { stageA(buf ^ 1, m + 1); stageB(buf ^ 1, m + 1); }

    // A frag in contiguous k = 16*h + s order, same as B (2x ds_load_b128)
    Frag fa;
    {
      const unsigned short* ap = &As[buf][(ts * 16 + l16) * AROW + (lhi ? 16 : 0)];
      fa.q[0] = *(const uint4*)ap;          // K 0..7   (hi-lanes: 16..23)
      fa.q[1] = *(const uint4*)(ap + 8);    // K 8..15  (hi-lanes: 24..31)
    }
    #pragma unroll
    for (int j = 0; j < 8; ++j) {
      const int os = og * 8 + j;
      Frag fb;
      const unsigned short* bp = &Bs[buf][(os * 16 + l16) * BROW + (lhi ? 16 : 0)];
      fb.q[0] = *(const uint4*)bp;          // K 0..7   (hi-lanes: 16..23)
      fb.q[1] = *(const uint4*)(bp + 8);    // K 8..15  (hi-lanes: 24..31)
      acc[j] = __builtin_amdgcn_wmma_f32_16x16x32_bf16(
          false, fa.v, false, fb.v, (short)0, acc[j], false, false);
    }
  }

  __syncthreads();
  // accumulators -> LDS transpose buffer Ct[o_local][t_local]
  #pragma unroll
  for (int j = 0; j < 8; ++j) {
    const int ol = (og * 8 + j) * 16 + l16;
    #pragma unroll
    for (int r = 0; r < 8; ++r) {
      const int tl = ts * 16 + r + (lhi ? 8 : 0);
      Ct[ol * (BT + 1) + tl] = acc[j][r];
    }
  }
  __syncthreads();

  // coalesced store: lanes walk t (contiguous in out[b,o,:]); add bias here
  const int s_t  = tid & 63;
  const int s_og = tid >> 6;
  const int tg   = t0 + s_t;
  if (tg < OUT_LEN) {
    #pragma unroll 4
    for (int g = 0; g < 64; ++g) {
      const int o = s_og * 64 + g;
      out[((size_t)b * COUT + o) * OUT_LEN + tg] = Ct[o * (BT + 1) + s_t] + bias[o];
    }
  }
}

extern "C" void kernel_launch(void* const* d_in, const int* in_sizes, int n_in,
                              void* d_out, int out_size, void* d_ws, size_t ws_size,
                              hipStream_t stream) {
  (void)in_sizes; (void)n_in; (void)out_size;
  const float* x       = (const float*)d_in[0];
  const float* offsets = (const float*)d_in[1];
  const float* weight  = (const float*)d_in[2];
  const float* bias    = (const float*)d_in[3];
  float* out = (float*)d_out;

  const int use_ws = (ws_size >= (size_t)WTILE_BYTES) ? 1 : 0;
  unsigned short* wtiles = (unsigned short*)d_ws;

  if (use_ws) {
    hipLaunchKernelGGL(convert_weight_tiles, dim3(96), dim3(256), 0, stream,
                       weight, wtiles);
  }
  hipLaunchKernelGGL(deform_conv1d_wmma, dim3((OUT_LEN + BT - 1) / BT, 1, 8),
                     dim3(256), 0, stream,
                     x, offsets, weight, bias, wtiles, use_ws, out);
}
